// ColorMemory_81509889343918
// MI455X (gfx1250) — matrix operand order repaired and checked by reading, not compiled, not run on hardware
//
#include <hip/hip_runtime.h>
#include <math.h>

typedef __bf16 bf16;
typedef __attribute__((ext_vector_type(16))) __bf16 v16bf;
typedef __attribute__((ext_vector_type(8)))  float  v8f;

typedef int v4i __attribute__((vector_size(16)));
typedef __attribute__((address_space(1))) v4i* gvp;   // global int4*
typedef __attribute__((address_space(3))) v4i* lvp;   // LDS int4*

#if defined(__AMDGCN__) && __has_builtin(__builtin_amdgcn_global_load_async_to_lds_b128)
#define HAVE_ASYNC_LDS 1
#else
#define HAVE_ASYNC_LDS 0
#endif

__device__ __forceinline__ void wait_async0() {
#if __has_builtin(__builtin_amdgcn_s_wait_asynccnt)
  __builtin_amdgcn_s_wait_asynccnt(0);
#else
  asm volatile("s_wait_asynccnt 0" ::: "memory");
#endif
}

__device__ __forceinline__ float wave_sum(float v) {
#pragma unroll
  for (int m = 16; m >= 1; m >>= 1) v += __shfl_xor(v, m, 32);
  return v;
}
__device__ __forceinline__ float wave_max(float v) {
#pragma unroll
  for (int m = 16; m >= 1; m >>= 1) v = fmaxf(v, __shfl_xor(v, m, 32));
  return v;
}

// ---------------------------------------------------------------------------
// K0: x[b,c,h,w] -> xt[b,hw,c] (fp32, raw) and xn = LN1(xt)*w+b (bf16)
// ---------------------------------------------------------------------------
__global__ __launch_bounds__(256) void ln1_transpose_kernel(
    const float* __restrict__ x, const float* __restrict__ w,
    const float* __restrict__ b, float* __restrict__ xt, bf16* __restrict__ xn)
{
  __shared__ float tile[32][257];
  const int tid = threadIdx.x, wave = tid >> 5, lane = tid & 31;
  const int chunk = blockIdx.x;          // 1024 blocks: 8 batches * 128 s-tiles
  const int bb = chunk >> 7;
  const int s0 = (chunk & 127) << 5;
#pragma unroll
  for (int it = 0; it < 32; ++it) {
    int c = it * 8 + wave;
    tile[lane][c] = x[(((size_t)bb * 256 + c) << 12) + s0 + lane];
  }
  __syncthreads();
#pragma unroll
  for (int rr = 0; rr < 4; ++rr) {
    int sl = wave * 4 + rr;
    float vals[8], s = 0.f, s2 = 0.f;
#pragma unroll
    for (int j = 0; j < 8; ++j) {
      float v = tile[sl][lane + j * 32];
      vals[j] = v; s += v; s2 += v * v;
    }
    s = wave_sum(s); s2 = wave_sum(s2);
    float mean = s * (1.f / 256.f);
    float var  = s2 * (1.f / 256.f) - mean * mean;
    float rstd = rsqrtf(var + 1e-5f);
    size_t row = (size_t)bb * 4096 + s0 + sl;
#pragma unroll
    for (int j = 0; j < 8; ++j) {
      int c = lane + j * 32;
      float v = vals[j];
      xt[row * 256 + c] = v;
      xn[row * 256 + c] = (bf16)((v - mean) * rstd * w[c] + b[c]);
    }
  }
}

// ---------------------------------------------------------------------------
// Softmax over 512-wide rows, fp32 in -> bf16 out. One wave per row.
// ---------------------------------------------------------------------------
__global__ __launch_bounds__(256) void softmax_kernel(
    const float* __restrict__ in, bf16* __restrict__ out)
{
  const int wave = threadIdx.x >> 5, lane = threadIdx.x & 31;
  const size_t row = (size_t)blockIdx.x * 8 + wave;
  const float* p = in + row * 512;
  float v[16], mx = -INFINITY;
#pragma unroll
  for (int j = 0; j < 16; ++j) { v[j] = p[lane + j * 32]; mx = fmaxf(mx, v[j]); }
  mx = wave_max(mx);
  float s = 0.f;
#pragma unroll
  for (int j = 0; j < 16; ++j) { v[j] = __expf(v[j] - mx); s += v[j]; }
  s = wave_sum(s);
  float inv = 1.f / s;
  bf16* q = out + row * 512;
#pragma unroll
  for (int j = 0; j < 16; ++j) q[lane + j * 32] = (bf16)(v[j] * inv);
}

// ---------------------------------------------------------------------------
// K5: y = LN2(concat(xt, color_prior)) -> ynf (fp32) and ynb (bf16, in-place
// over color_prior buffer: each wave reads its full row before storing).
// ---------------------------------------------------------------------------
__global__ __launch_bounds__(256) void concat_ln_kernel(
    const float* __restrict__ xt, const float* __restrict__ cp,
    const float* __restrict__ w, const float* __restrict__ b,
    float* __restrict__ ynf, bf16* __restrict__ ynb)
{
  const int wave = threadIdx.x >> 5, lane = threadIdx.x & 31;
  const size_t row = (size_t)blockIdx.x * 8 + wave;
  float v[16], s = 0.f, s2 = 0.f;
#pragma unroll
  for (int j = 0; j < 16; ++j) {
    int c = lane + j * 32;
    float t = (j < 8) ? xt[row * 256 + c] : cp[row * 256 + (c - 256)];
    v[j] = t; s += t; s2 += t * t;
  }
  s = wave_sum(s); s2 = wave_sum(s2);
  float mean = s * (1.f / 512.f);
  float var  = s2 * (1.f / 512.f) - mean * mean;
  float rstd = rsqrtf(var + 1e-5f);
#pragma unroll
  for (int j = 0; j < 16; ++j) {
    int c = lane + j * 32;
    float t = (v[j] - mean) * rstd * w[c] + b[c];
    ynf[row * 512 + c] = t;
    ynb[row * 512 + c] = (bf16)t;
  }
}

// K8: z = LN3(ynf + mlp) -> bf16
__global__ __launch_bounds__(256) void add_ln_kernel(
    const float* __restrict__ a, const float* __restrict__ m,
    const float* __restrict__ w, const float* __restrict__ b,
    bf16* __restrict__ z)
{
  const int wave = threadIdx.x >> 5, lane = threadIdx.x & 31;
  const size_t row = (size_t)blockIdx.x * 8 + wave;
  float v[16], s = 0.f, s2 = 0.f;
#pragma unroll
  for (int j = 0; j < 16; ++j) {
    int c = lane + j * 32;
    float t = a[row * 512 + c] + m[row * 512 + c];
    v[j] = t; s += t; s2 += t * t;
  }
  s = wave_sum(s); s2 = wave_sum(s2);
  float mean = s * (1.f / 512.f);
  float var  = s2 * (1.f / 512.f) - mean * mean;
  float rstd = rsqrtf(var + 1e-5f);
#pragma unroll
  for (int j = 0; j < 16; ++j) {
    int c = lane + j * 32;
    z[row * 512 + c] = (bf16)((v[j] - mean) * rstd * w[c] + b[c]);
  }
}

// ---------------------------------------------------------------------------
// Small prep kernels
// ---------------------------------------------------------------------------
__global__ void cvt_kernel(const float* __restrict__ src, bf16* __restrict__ dst, int n) {
  int i = blockIdx.x * 256 + threadIdx.x;
  if (i < n) dst[i] = (bf16)src[i];
}

__global__ void transpose_cvt_kernel(const float* __restrict__ src,
                                     bf16* __restrict__ dst, int R, int C) {
  int i = blockIdx.x * 256 + threadIdx.x;
  if (i >= R * C) return;
  int r = i / C, c = i % C;
  dst[(size_t)c * R + r] = (bf16)src[i];
}

// ab[i,n,f] = concat(a_embed[cc[i,n,0]], b_embed[cc[i,n,1]]) -> bf16 [4,512,512]
__global__ void gather_ab_kernel(const int* __restrict__ cc,
                                 const float* __restrict__ ae,
                                 const float* __restrict__ be,
                                 bf16* __restrict__ ab) {
  int i = blockIdx.x * 256 + threadIdx.x;   // 4*512*512
  int f = i & 511, n = (i >> 9) & 511, h = i >> 18;
  float v;
  if (f < 256) { int idx = cc[(h * 512 + n) * 2 + 0]; v = ae[idx * 256 + f]; }
  else         { int idx = cc[(h * 512 + n) * 2 + 1]; v = be[idx * 256 + (f - 256)]; }
  ab[i] = (bf16)v;
}

// b_qs[n] = q_b . sem[n,:]
__global__ __launch_bounds__(256) void bqs_kernel(const float* __restrict__ qb,
    const float* __restrict__ sem, float* __restrict__ bqs) {
  int wave = threadIdx.x >> 5, lane = threadIdx.x & 31;
  int n = blockIdx.x * 8 + wave;
  float s = 0.f;
  for (int j = lane; j < 768; j += 32) s += qb[j] * sem[n * 768 + j];
  s = wave_sum(s);
  if (lane == 0) bqs[n] = s;
}

// colorT[b,d,n] = sum_i cls[b,i] * ce[i,n,d]   (ce already includes ce_b)
__global__ void colort_kernel(const float* __restrict__ cls,
                              const float* __restrict__ ce,
                              bf16* __restrict__ out) {
  int i = blockIdx.x * 256 + threadIdx.x;   // 8*256*512
  int n = i & 511, d = (i >> 9) & 255, bb = i >> 17;
  float s = 0.f;
#pragma unroll
  for (int k = 0; k < 4; ++k)
    s += cls[bb * 4 + k] * ce[((size_t)(k * 512 + n)) * 256 + d];
  out[i] = (bf16)s;
}

// ---------------------------------------------------------------------------
// WMMA GEMM: C[M,N] = A[M,K] * B[N,K]^T (+bias[N])
// A row-major bf16 (lda), B stored transposed [N,K] row-major bf16 (ldb).
// 256 threads = 8 waves (2x4); macro tile 128x256; wave tile 64x64; BK=32.
// Double-buffered LDS with CDNA5 async global->LDS copies (ASYNCcnt) when
// available; falls back to load/ds_store staging otherwise.
// EPI: 0 = fp32 store, 1 = bf16 store, 2 = GELU->bf16, 3 = fp32 scatter to NCHW
// ---------------------------------------------------------------------------
template <int EPI>
__global__ __launch_bounds__(256) void gemm_bf16_kernel(
    const bf16* __restrict__ A, int lda, long long strideA,
    const bf16* __restrict__ B, int ldb, long long strideB,
    const float* __restrict__ bias, long long strideBias,
    float* __restrict__ Cf, bf16* __restrict__ Cb, int ldc, long long strideC,
    int K)
{
  constexpr int BM = 128, BN = 256, BK = 32, AST = 40;  // +8 bf16 pad per row
  __shared__ __align__(16) bf16 As[2][BM * AST];
  __shared__ __align__(16) bf16 Bs[2][BN * AST];
  const int tid = threadIdx.x;
  const int wave = tid >> 5, lane = tid & 31;
  const int wm = (wave >> 2) * 64, wn = (wave & 3) * 64;
  const int m0 = blockIdx.y * BM, n0 = blockIdx.x * BN;
  const int z = blockIdx.z;
  A += (size_t)z * strideA;
  B += (size_t)z * strideB;
  if (bias) bias += (size_t)z * strideBias;

  v8f acc[4][4];
#pragma unroll
  for (int i = 0; i < 4; ++i)
#pragma unroll
    for (int j = 0; j < 4; ++j)
#pragma unroll
      for (int r = 0; r < 8; ++r) acc[i][j][r] = 0.f;

  const int lhalf = (lane >> 4) & 1;
  const int l15 = lane & 15;
  const int ar = tid >> 2;            // staging row 0..63
  const int akc = (tid & 3) * 8;      // staging k-chunk (8 bf16 = 16B)

  auto stage = [&](int kt, int pb) {
#if HAVE_ASYNC_LDS
#pragma unroll
    for (int it = 0; it < 2; ++it) {
      int r = ar + it * 64;
      const bf16* g = A + (size_t)(m0 + r) * lda + kt + akc;
      const bf16* l = &As[pb][r * AST + akc];
      __builtin_amdgcn_global_load_async_to_lds_b128(
          (gvp)(uintptr_t)g, (lvp)(unsigned)(uintptr_t)l, 0, 0);
    }
#pragma unroll
    for (int it = 0; it < 4; ++it) {
      int r = ar + it * 64;
      const bf16* g = B + (size_t)(n0 + r) * ldb + kt + akc;
      const bf16* l = &Bs[pb][r * AST + akc];
      __builtin_amdgcn_global_load_async_to_lds_b128(
          (gvp)(uintptr_t)g, (lvp)(unsigned)(uintptr_t)l, 0, 0);
    }
#else
#pragma unroll
    for (int it = 0; it < 2; ++it) {
      int r = ar + it * 64;
      *reinterpret_cast<uint4*>(&As[pb][r * AST + akc]) =
          *reinterpret_cast<const uint4*>(A + (size_t)(m0 + r) * lda + kt + akc);
    }
#pragma unroll
    for (int it = 0; it < 4; ++it) {
      int r = ar + it * 64;
      *reinterpret_cast<uint4*>(&Bs[pb][r * AST + akc]) =
          *reinterpret_cast<const uint4*>(B + (size_t)(n0 + r) * ldb + kt + akc);
    }
#endif
  };

  stage(0, 0);
  const int nk = K / BK;
  for (int i = 0; i < nk; ++i) {
    const int pb = i & 1;
#if HAVE_ASYNC_LDS
    wait_async0();                 // buffer pb fully deposited in LDS
#endif
    __syncthreads();               // visible to all waves
    if (i + 1 < nk) stage((i + 1) * BK, pb ^ 1);   // overlap with compute

    // A fragments per ISA layout: lane<16 holds K{0..7,16..23}, lane>=16 K{8..15,24..31}
    union { uint4 q[2]; v16bf v; } af[4];
#pragma unroll
    for (int mi = 0; mi < 4; ++mi) {
      int row = wm + mi * 16 + l15;
      int k0 = lhalf * 8;
      af[mi].q[0] = *reinterpret_cast<const uint4*>(&As[pb][row * AST + k0]);
      af[mi].q[1] = *reinterpret_cast<const uint4*>(&As[pb][row * AST + k0 + 16]);
    }
#pragma unroll
    for (int ni = 0; ni < 4; ++ni) {
      // B fragment: lane<16 -> N=l, K{0..7,16..23}; lane>=16 -> N=l-16, K{8..15,24..31}
      union { uint4 q[2]; v16bf v; } bF;
      int nrow = wn + ni * 16 + l15;
      int kb = lhalf * 8;
      bF.q[0] = *reinterpret_cast<const uint4*>(&Bs[pb][nrow * AST + kb]);
      bF.q[1] = *reinterpret_cast<const uint4*>(&Bs[pb][nrow * AST + kb + 16]);
#pragma unroll
      for (int mi = 0; mi < 4; ++mi)
        acc[mi][ni] = __builtin_amdgcn_wmma_f32_16x16x32_bf16(
            false, af[mi].v, false, bF.v, (short)0, acc[mi][ni], false, false);
    }
    __syncthreads();               // all waves done reading pb before overwrite
  }

  // epilogue: C layout — VGPR r, lane l: M = r + 8*(l>>4), N = l&15
#pragma unroll
  for (int mi = 0; mi < 4; ++mi) {
#pragma unroll
    for (int ni = 0; ni < 4; ++ni) {
#pragma unroll
      for (int r = 0; r < 8; ++r) {
        int row = m0 + wm + mi * 16 + lhalf * 8 + r;
        int col = n0 + wn + ni * 16 + l15;
        float v = acc[mi][ni][r];
        if (bias) v += bias[col];
        if constexpr (EPI == 0) {
          Cf[(size_t)z * strideC + (size_t)row * ldc + col] = v;
        } else if constexpr (EPI == 1) {
          Cb[(size_t)z * strideC + (size_t)row * ldc + col] = (bf16)v;
        } else if constexpr (EPI == 2) {
          float g = 0.5f * v * (1.0f + erff(v * 0.70710678118654752f));
          Cb[(size_t)z * strideC + (size_t)row * ldc + col] = (bf16)g;
        } else {  // EPI==3: scatter to out[b, col, s] with row = b*4096 + s
          int s = row & 4095, bidx = row >> 12;
          Cf[((size_t)bidx * 256 + col) * 4096 + s] = v;
        }
      }
    }
  }
}

// ---------------------------------------------------------------------------
extern "C" void kernel_launch(void* const* d_in, const int* in_sizes, int n_in,
                              void* d_out, int out_size, void* d_ws, size_t ws_size,
                              hipStream_t stream) {
  const float* x    = (const float*)d_in[0];
  const float* cls  = (const float*)d_in[1];
  const int*   cc   = (const int*)d_in[2];
  const float* scen = (const float*)d_in[3];
  const float* ae   = (const float*)d_in[4];
  const float* be   = (const float*)d_in[5];
  const float* ce_w = (const float*)d_in[6];
  const float* ce_b = (const float*)d_in[7];
  const float* sem_w= (const float*)d_in[8];
  const float* sem_b= (const float*)d_in[9];
  const float* q_w  = (const float*)d_in[10];
  const float* q_b  = (const float*)d_in[11];
  const float* n1w  = (const float*)d_in[12];
  const float* n1b  = (const float*)d_in[13];
  const float* n2w  = (const float*)d_in[14];
  const float* n2b  = (const float*)d_in[15];
  const float* n3w  = (const float*)d_in[16];
  const float* n3b  = (const float*)d_in[17];
  const float* fc1w = (const float*)d_in[18];
  const float* fc1b = (const float*)d_in[19];
  const float* fc2w = (const float*)d_in[20];
  const float* fc2b = (const float*)d_in[21];
  const float* convw= (const float*)d_in[22];
  const float* convb= (const float*)d_in[23];
  float* out = (float*)d_out;
  (void)in_sizes; (void)n_in; (void)out_size; (void)ws_size;

  char* ws = (char*)d_ws;
  size_t off = 0;
  auto alloc = [&](size_t bytes) {
    char* p = ws + off;
    off += (bytes + 255) & ~(size_t)255;
    return p;
  };
  float* buf1   = (float*)alloc(67108864);  // logits, later mlp (32768x512 f32)
  float* xt     = (float*)alloc(33554432);  // 32768x256 f32
  bf16*  buf3   = (bf16*) alloc(33554432);  // xn (32768x256), later z (32768x512)
  bf16*  buf4   = (bf16*) alloc(33554432);  // attn, later h1 (32768x512)
  float* cp     = (float*)alloc(33554432);  // color_prior f32; reused as ynb bf16
  float* ynf    = (float*)alloc(67108864);  // 32768x512 f32
  bf16*  sc_bf  = (bf16*) alloc(524288);
  bf16*  semwT  = (bf16*) alloc(786432);
  float* sem_f  = (float*)alloc(1572864);
  bf16*  sem_bf = (bf16*) alloc(786432);
  bf16*  qw_bf  = (bf16*) alloc(393216);
  bf16*  wqsT   = (bf16*) alloc(262144);
  float* bqs    = (float*)alloc(2048);
  bf16*  ab_bf  = (bf16*) alloc(2097152);
  bf16*  cewT   = (bf16*) alloc(1048576);
  float* ce_f   = (float*)alloc(2097152);
  bf16*  colorT = (bf16*) alloc(2097152);
  bf16*  fc1T   = (bf16*) alloc(524288);
  bf16*  fc2T   = (bf16*) alloc(524288);
  bf16*  convT  = (bf16*) alloc(262144);
  bf16* xn  = buf3;
  bf16* z   = buf3;
  bf16* ynb = (bf16*)cp;

  // ---- small prep ----
  cvt_kernel<<<(512 * 512 + 255) / 256, 256, 0, stream>>>(scen, sc_bf, 512 * 512);
  transpose_cvt_kernel<<<(512 * 768 + 255) / 256, 256, 0, stream>>>(sem_w, semwT, 512, 768);
  cvt_kernel<<<(256 * 768 + 255) / 256, 256, 0, stream>>>(q_w, qw_bf, 256 * 768);
  for (int i = 0; i < 4; ++i)
    transpose_cvt_kernel<<<(512 * 256 + 255) / 256, 256, 0, stream>>>(
        ce_w + (size_t)i * 512 * 256, cewT + (size_t)i * 256 * 512, 512, 256);
  transpose_cvt_kernel<<<(512 * 512 + 255) / 256, 256, 0, stream>>>(fc1w, fc1T, 512, 512);
  transpose_cvt_kernel<<<(512 * 512 + 255) / 256, 256, 0, stream>>>(fc2w, fc2T, 512, 512);
  transpose_cvt_kernel<<<(512 * 256 + 255) / 256, 256, 0, stream>>>(convw, convT, 512, 256);
  gather_ab_kernel<<<4096, 256, 0, stream>>>(cc, ae, be, ab_bf);

  // sem = semantic_centers @ sem_w + sem_b   [512,768]
  gemm_bf16_kernel<0><<<dim3(3, 4, 1), 256, 0, stream>>>(
      sc_bf, 512, 0, semwT, 512, 0, sem_b, 0, sem_f, nullptr, 768, 0, 512);
  cvt_kernel<<<(512 * 768 + 255) / 256, 256, 0, stream>>>(sem_f, sem_bf, 512 * 768);
  // W_qs^T[n,c] = sem[n,:] . q_w[c,:]       [512,256] bf16
  gemm_bf16_kernel<1><<<dim3(1, 4, 1), 256, 0, stream>>>(
      sem_bf, 768, 0, qw_bf, 768, 0, nullptr, 0, nullptr, wqsT, 256, 0, 768);
  bqs_kernel<<<64, 256, 0, stream>>>(q_b, sem_f, bqs);
  // ce[i] = ab[i] @ ce_w[i] + ce_b[i]       [4,512,256]
  gemm_bf16_kernel<0><<<dim3(1, 4, 4), 256, 0, stream>>>(
      ab_bf, 512, (long long)512 * 512, cewT, 512, (long long)256 * 512,
      ce_b, 256, ce_f, nullptr, 256, (long long)512 * 256, 512);
  colort_kernel<<<(8 * 256 * 512) / 256, 256, 0, stream>>>(cls, ce_f, colorT);

  // ---- main path ----
  ln1_transpose_kernel<<<1024, 256, 0, stream>>>(x, n1w, n1b, xt, xn);
  // logits = xn @ W_qs + b_qs               [32768,512]
  gemm_bf16_kernel<0><<<dim3(2, 256, 1), 256, 0, stream>>>(
      xn, 256, 0, wqsT, 256, 0, bqs, 0, buf1, nullptr, 512, 0, 256);
  softmax_kernel<<<4096, 256, 0, stream>>>(buf1, buf4);
  // color_prior[b] = attn[b] @ color_embed[b]   [8][4096,256]
  gemm_bf16_kernel<0><<<dim3(1, 32, 8), 256, 0, stream>>>(
      buf4, 512, (long long)4096 * 512, colorT, 512, (long long)256 * 512,
      nullptr, 0, cp, nullptr, 256, (long long)4096 * 256, 512);
  concat_ln_kernel<<<4096, 256, 0, stream>>>(xt, cp, n2w, n2b, ynf, ynb);
  // h1 = gelu(yn @ fc1 + b)                  [32768,512] bf16
  gemm_bf16_kernel<2><<<dim3(2, 256, 1), 256, 0, stream>>>(
      ynb, 512, 0, fc1T, 512, 0, fc1b, 0, nullptr, buf4, 512, 0, 512);
  // mlp = h1 @ fc2 + b                       [32768,512] f32
  gemm_bf16_kernel<0><<<dim3(2, 256, 1), 256, 0, stream>>>(
      buf4, 512, 0, fc2T, 512, 0, fc2b, 0, buf1, nullptr, 512, 0, 512);
  add_ln_kernel<<<4096, 256, 0, stream>>>(ynf, buf1, n3w, n3b, z);
  // out = z @ conv + b, scattered to [b,c,h,w]
  gemm_bf16_kernel<3><<<dim3(1, 256, 1), 256, 0, stream>>>(
      z, 512, 0, convT, 512, 0, convb, 0, out, nullptr, 0, 0, 512);
}
